// AdaptiveInputSoftmax_60567628808647
// MI455X (gfx1250) — matrix operand order repaired and checked by reading, not compiled, not run on hardware
//
#include <hip/hip_runtime.h>

typedef __bf16 bf16;
typedef __attribute__((ext_vector_type(16))) __bf16 v16bf;
typedef __attribute__((ext_vector_type(8)))  __bf16 v8bf;
typedef __attribute__((ext_vector_type(8)))  float   v8f;
typedef __attribute__((ext_vector_type(4)))  float   v4f;

#define TOK   2048      // N*T tokens
#define HDIM  512
#define VOCAB 100000
#define CUT0  20000
#define CUT1  50000
#define NEG_BIG (-3.0e38f)

// ---------------------------------------------------------------------------
// f32 -> bf16 elementwise
__global__ void cvt_x_kernel(const float* __restrict__ x, bf16* __restrict__ o, long n) {
    long i = (long)blockIdx.x * blockDim.x + threadIdx.x;
    if (i < n) o[i] = (bf16)x[i];
}

// W[K][N] f32 -> Bt[Npad][K] bf16 (transpose + pad rows >= N with zero).
__global__ void cvt_transpose_kernel(const float* __restrict__ W, bf16* __restrict__ Bt,
                                     int K, int N, int Npad) {
    long total = (long)K * Npad;
    long i = (long)blockIdx.x * blockDim.x + threadIdx.x;
    if (i >= total) return;
    int n = (int)(i % Npad);
    int k = (int)(i / Npad);
    float v = (n < N) ? W[(long)k * N + n] : 0.0f;
    Bt[(long)n * K + k] = (bf16)v;
}

// ---------------------------------------------------------------------------
__device__ inline void stat_combine(float& m, float& s, float m2, float s2) {
    float mn = fmaxf(m, m2);
    s = s * __expf(m - mn) + s2 * __expf(m2 - mn);
    m = mn;
}

// Fragment pair for one 32-wide K step of a 16x64 output strip.
struct Frag {
    v8bf a0, a1;
    v8bf b0[4], b1[4];
};

template <int K>
__device__ __forceinline__ void load_frag(Frag& f, const bf16* __restrict__ aRow,
                                          const bf16* __restrict__ bCol, int k) {
    f.a0 = *(const v8bf*)(aRow + k);
    f.a1 = *(const v8bf*)(aRow + k + 8);
#pragma unroll
    for (int t = 0; t < 4; ++t) {
        const bf16* bp = bCol + t * 16 * K + k;
        f.b0[t] = *(const v8bf*)(bp);
        f.b1[t] = *(const v8bf*)(bp + 8);
    }
}

__device__ __forceinline__ void do_wmma(v8f acc[4], const Frag& f) {
    union { v16bf v; v8bf h[2]; } a;
    a.h[0] = f.a0; a.h[1] = f.a1;
#pragma unroll
    for (int t = 0; t < 4; ++t) {
        union { v16bf v; v8bf h[2]; } b;
        b.h[0] = f.b0[t]; b.h[1] = f.b1[t];
        acc[t] = __builtin_amdgcn_wmma_f32_16x16x32_bf16(
            false, a.v, false, b.v, (short)0, acc[t], false, false);
    }
}

// WMMA GEMM: C[M,N] = A[M,K]*B[K,N]; A row-major bf16, Bt = B^T row-major bf16.
// Wave computes a 16x64 strip (4 accumulators); ping-pong pipelined K loop.
// MODE 0: bf16 output (row stride ldH)
// MODE 1: f32 logits (nontemporal) + fused online-softmax partial stats
// MODE 2: MODE 1 + cols >= splitN routed to extras[row*2 + (col-splitN)]
template <int K, int MODE>
__global__ __launch_bounds__(256)
void gemm_wmma_kernel(const bf16* __restrict__ A, const bf16* __restrict__ Bt, int N,
                      float* __restrict__ outF, long ldOut, long colOff,
                      bf16* __restrict__ outH, int ldH,
                      float* __restrict__ extras, int splitN,
                      float2* __restrict__ statsPartial, int ldP) {
    const int wave = threadIdx.x >> 5;
    const int lane = threadIdx.x & 31;
    const int m0   = blockIdx.y * 16;
    const int n0   = blockIdx.x * 512 + wave * 64;
    if (n0 >= N) return;                       // wave-uniform exit

    const int half = lane >> 4;                // 0: lanes 0-15, 1: lanes 16-31
    const int l16  = lane & 15;

    v8f acc[4];
    acc[0] = v8f{}; acc[1] = v8f{}; acc[2] = v8f{}; acc[3] = v8f{};

    // A fragment: row m0+l16; half 0 -> K 0..15, half 1 -> K 16..31
    const bf16* aRow = A + (long)(m0 + l16) * K + half * 16;
    // B fragment: column n0+l16 (+16 per tile); half 0 -> K 0..15, half 1 -> K 16..31
    const bf16* bCol = Bt + (long)(n0 + l16) * K + half * 16;

    // ---- pipelined K loop: ping-pong buffers, no register rotation ----
    Frag f0, f1;
    load_frag<K>(f0, aRow, bCol, 0);
    if constexpr (K == 32) {
        do_wmma(acc, f0);
    } else {
        int k0 = 0;
#pragma unroll
        for (; k0 + 64 < K; k0 += 64) {
            load_frag<K>(f1, aRow, bCol, k0 + 32);
            do_wmma(acc, f0);
            load_frag<K>(f0, aRow, bCol, k0 + 64);
            do_wmma(acc, f1);
        }
        load_frag<K>(f1, aRow, bCol, k0 + 32);
        do_wmma(acc, f0);
        do_wmma(acc, f1);
    }

    // ---- store: VGPR i -> M = half*8 + i, N = l16 (per 16x16 tile) ----
#pragma unroll
    for (int t = 0; t < 4; ++t) {
        int col = n0 + t * 16 + l16;
        if (col >= N) continue;
#pragma unroll
        for (int i = 0; i < 8; ++i) {
            int   row = m0 + half * 8 + i;
            float v   = acc[t][i];
            if constexpr (MODE == 0) {
                outH[(long)row * ldH + col] = (bf16)v;
            } else if constexpr (MODE == 2) {
                if (col >= splitN) extras[row * 2 + (col - splitN)] = v;
                else __builtin_nontemporal_store(v, outF + (long)row * ldOut + colOff + col);
            } else {
                __builtin_nontemporal_store(v, outF + (long)row * ldOut + colOff + col);
            }
        }
    }

    // ---- fused online-softmax partial stats over this wave's 64 columns ----
    if constexpr (MODE != 0) {
#pragma unroll
        for (int i = 0; i < 8; ++i) {
            float m = NEG_BIG, s = 0.0f;
#pragma unroll
            for (int t = 0; t < 4; ++t) {
                int col = n0 + t * 16 + l16;
                float v = (col < N) ? acc[t][i] : NEG_BIG;
                if (v > m) { s = s * __expf(m - v) + 1.0f; m = v; }
                else       { s += __expf(v - m); }
            }
            // reduce across the 16 lanes of this half (xor 1,2,4,8 stays in-half)
#pragma unroll
            for (int off = 1; off < 16; off <<= 1) {
                float m2 = __shfl_xor(m, off, 32);
                float s2 = __shfl_xor(s, off, 32);
                stat_combine(m, s, m2, s2);
            }
            if (l16 == 0) {
                int row  = m0 + half * 8 + i;
                int wIdx = blockIdx.x * 8 + wave;
                statsPartial[(long)row * ldP + wIdx] = make_float2(m, s);
            }
        }
    }
}

// ---------------------------------------------------------------------------
// Fold per-wave-strip partials into one (max, sum) per row. One wave per row.
// Slot w valid iff its strip start n0 = (w>>3)*512 + (w&7)*64 < N.
__global__ __launch_bounds__(32)
void combine_stats_kernel(const float2* __restrict__ partials, int ldP,
                          int W, int N, float2* __restrict__ stats) {
    int row  = blockIdx.x;
    int lane = threadIdx.x;
    float m = NEG_BIG, s = 0.0f;
    for (int w = lane; w < W; w += 32) {
        int n0 = ((w >> 3) << 9) + ((w & 7) << 6);
        if (n0 >= N) continue;
        float2 p = partials[(long)row * ldP + w];
        stat_combine(m, s, p.x, p.y);
    }
#pragma unroll
    for (int off = 16; off > 0; off >>= 1) {
        float m2 = __shfl_xor(m, off, 32);
        float s2 = __shfl_xor(s, off, 32);
        stat_combine(m, s, m2, s2);
    }
    if (lane == 0) stats[row] = make_float2(m, s);
}

// Head: normalize first CUT0 probs in-place (vec4); emit cluster probs.
__global__ __launch_bounds__(256)
void head_norm_kernel(float* __restrict__ base, const float* __restrict__ extras,
                      const float2* __restrict__ stats, float* __restrict__ cluster) {
    int row = blockIdx.x;
    float2 st = stats[row];
    float inv = 1.0f / st.y;
    v4f* p = (v4f*)(base + (long)row * VOCAB);
    for (int i = threadIdx.x; i < CUT0 / 4; i += 256) {
        v4f v = __builtin_nontemporal_load(p + i);
        v4f r;
        r.x = __expf(v.x - st.x) * inv;
        r.y = __expf(v.y - st.x) * inv;
        r.z = __expf(v.z - st.x) * inv;
        r.w = __expf(v.w - st.x) * inv;
        __builtin_nontemporal_store(r, p + i);
    }
    if (threadIdx.x < 2)
        cluster[row * 2 + threadIdx.x] = __expf(extras[row * 2 + threadIdx.x] - st.x) * inv;
}

// Tail: normalize in-place scaled by cluster probability (vec4).
__global__ __launch_bounds__(256)
void tail_norm_kernel(float* __restrict__ base, long ldRow, int n4,
                      const float2* __restrict__ stats,
                      const float* __restrict__ cluster, int cidx) {
    int row = blockIdx.x;
    float2 st = stats[row];
    float scale = cluster[row * 2 + cidx] / st.y;
    v4f* p = (v4f*)(base + (long)row * ldRow);
    for (int i = threadIdx.x; i < n4; i += 256) {
        v4f v = __builtin_nontemporal_load(p + i);
        v4f r;
        r.x = __expf(v.x - st.x) * scale;
        r.y = __expf(v.y - st.x) * scale;
        r.z = __expf(v.z - st.x) * scale;
        r.w = __expf(v.w - st.x) * scale;
        __builtin_nontemporal_store(r, p + i);
    }
}

// ---------------------------------------------------------------------------
static inline long cdivl(long a, long b) { return (a + b - 1) / b; }

extern "C" void kernel_launch(void* const* d_in, const int* in_sizes, int n_in,
                              void* d_out, int out_size, void* d_ws, size_t ws_size,
                              hipStream_t stream) {
    const float* x  = (const float*)d_in[0];   // [2048, 512]
    const float* hw = (const float*)d_in[1];   // [512, 20002]
    const float* p0 = (const float*)d_in[2];   // [512, 128]
    const float* w0 = (const float*)d_in[3];   // [128, 30000]
    const float* p1 = (const float*)d_in[4];   // [512, 32]
    const float* w1 = (const float*)d_in[5];   // [32, 50000]
    float* out = (float*)d_out;

    const int N_HEAD = CUT0 + 2;            // 20002
    const int NP_HEAD = 20032;              // pad to multiple of 64
    const int N_T0 = CUT1 - CUT0;           // 30000
    const int NP_T0 = 30016;
    const int N_T1 = VOCAB - CUT1;          // 50000
    const int NP_T1 = 50048;
    const int NB_HEAD = 40, NB_T0 = 59, NB_T1 = 98;     // ceil(N/512)
    const int W_HEAD = NB_HEAD * 8, W_T0 = NB_T0 * 8, W_T1 = NB_T1 * 8;

    char* ws = (char*)d_ws;
    size_t off = 0;
    auto alloc = [&](size_t bytes) -> void* {
        void* p = ws + off;
        off = (off + bytes + 255) & ~(size_t)255;
        return p;
    };
    bf16*   xb     = (bf16*)  alloc((size_t)TOK * HDIM * 2);
    bf16*   hwT    = (bf16*)  alloc((size_t)NP_HEAD * HDIM * 2);
    bf16*   p0T    = (bf16*)  alloc((size_t)128 * HDIM * 2);
    bf16*   w0T    = (bf16*)  alloc((size_t)NP_T0 * 128 * 2);
    bf16*   p1T    = (bf16*)  alloc((size_t)64 * HDIM * 2);
    bf16*   w1T    = (bf16*)  alloc((size_t)NP_T1 * 32 * 2);
    bf16*   xp0    = (bf16*)  alloc((size_t)TOK * 128 * 2);
    bf16*   xp1    = (bf16*)  alloc((size_t)TOK * 32 * 2);
    float*  extras = (float*) alloc((size_t)TOK * 2 * 4);
    float*  clus   = (float*) alloc((size_t)TOK * 2 * 4);
    float2* stH    = (float2*)alloc((size_t)TOK * sizeof(float2));
    float2* st0    = (float2*)alloc((size_t)TOK * sizeof(float2));
    float2* st1    = (float2*)alloc((size_t)TOK * sizeof(float2));
    float2* pH     = (float2*)alloc((size_t)TOK * W_HEAD * sizeof(float2));
    float2* pT0    = (float2*)alloc((size_t)TOK * W_T0  * sizeof(float2));
    float2* pT1    = (float2*)alloc((size_t)TOK * W_T1  * sizeof(float2));
    (void)ws_size; (void)in_sizes; (void)n_in; (void)out_size;

    // --- conversions ---
    {
        long n = (long)TOK * HDIM;
        cvt_x_kernel<<<cdivl(n, 256), 256, 0, stream>>>(x, xb, n);
    }
    cvt_transpose_kernel<<<cdivl((long)HDIM * NP_HEAD, 256), 256, 0, stream>>>(hw, hwT, HDIM, N_HEAD, NP_HEAD);
    cvt_transpose_kernel<<<cdivl((long)HDIM * 128, 256),     256, 0, stream>>>(p0, p0T, HDIM, 128, 128);
    cvt_transpose_kernel<<<cdivl((long)128 * NP_T0, 256),    256, 0, stream>>>(w0, w0T, 128, N_T0, NP_T0);
    cvt_transpose_kernel<<<cdivl((long)HDIM * 64, 256),      256, 0, stream>>>(p1, p1T, HDIM, 32, 64);
    cvt_transpose_kernel<<<cdivl((long)32 * NP_T1, 256),     256, 0, stream>>>(w1, w1T, 32, N_T1, NP_T1);

    const dim3 blk(256);
    // --- projections (bf16 out, no stats) ---
    gemm_wmma_kernel<HDIM, 0><<<dim3(1, TOK / 16), blk, 0, stream>>>(
        xb, p0T, 128, nullptr, 0, 0, xp0, 128, nullptr, 0, nullptr, 0);
    gemm_wmma_kernel<HDIM, 0><<<dim3(1, TOK / 16), blk, 0, stream>>>(
        xb, p1T, 32, nullptr, 0, 0, xp1, 32, nullptr, 0, nullptr, 0);

    // --- logits with fused softmax partial stats ---
    gemm_wmma_kernel<HDIM, 2><<<dim3(NB_HEAD, TOK / 16), blk, 0, stream>>>(
        xb, hwT, N_HEAD, out, VOCAB, 0, nullptr, 0, extras, CUT0, pH, W_HEAD);
    gemm_wmma_kernel<128, 1><<<dim3(NB_T0, TOK / 16), blk, 0, stream>>>(
        xp0, w0T, N_T0, out, VOCAB, CUT0, nullptr, 0, nullptr, 0, pT0, W_T0);
    gemm_wmma_kernel<32, 1><<<dim3(NB_T1, TOK / 16), blk, 0, stream>>>(
        xp1, w1T, N_T1, out, VOCAB, CUT1, nullptr, 0, nullptr, 0, pT1, W_T1);

    // --- fold partials -> per-row (max, sum) ---
    combine_stats_kernel<<<TOK, 32, 0, stream>>>(pH,  W_HEAD, W_HEAD, N_HEAD, stH);
    combine_stats_kernel<<<TOK, 32, 0, stream>>>(pT0, W_T0,  W_T0,  N_T0,  st0);
    combine_stats_kernel<<<TOK, 32, 0, stream>>>(pT1, W_T1,  W_T1,  N_T1,  st1);

    // --- normalize (streaming, vec4, nontemporal) ---
    head_norm_kernel<<<TOK, blk, 0, stream>>>(out, extras, stH, clus);
    tail_norm_kernel<<<TOK, blk, 0, stream>>>(out + CUT0, VOCAB, N_T0 / 4, st0, clus, 0);
    tail_norm_kernel<<<TOK, blk, 0, stream>>>(out + CUT1, VOCAB, N_T1 / 4, st1, clus, 1);
}
